// MyDecoderLayerDAEFormer_2551210574447
// MI455X (gfx1250) — matrix operand order repaired and checked by reading, not compiled, not run on hardware
//
#include <hip/hip_runtime.h>
#include <hip/hip_bf16.h>

typedef __attribute__((ext_vector_type(16))) _Float16 v16h;
typedef __attribute__((ext_vector_type(4)))  _Float16 v4h;
typedef __attribute__((ext_vector_type(8)))  float    v8f;

#define CC   256
#define NT   4096
#define BB   8
#define C2C  1024
#define MTOT (BB * NT)          // 32768 rows

__device__ __forceinline__ float gelu_f(float x) {
    return 0.5f * x * (1.0f + erff(x * 0.70710678118654752f));
}
__device__ __forceinline__ float wave_sum(float v) {
    #pragma unroll
    for (int off = 16; off > 0; off >>= 1) v += __shfl_xor(v, off, 32);
    return v;
}
__device__ __forceinline__ float wave_max(float v) {
    #pragma unroll
    for (int off = 16; off > 0; off >>= 1) v = fmaxf(v, __shfl_xor(v, off, 32));
    return v;
}

// ---------------------------------------------------------------------------
// Generic strided, batched WMMA GEMM:
//   Out[m,n] (+= resid) = act( sum_k A[m,k] * B[n,k] + bias[n] )
// Tile: 128(M) x 64(N) per 256-thread block; 8 waves, each wave 16x64.
// K stepped by 32 through double-buffered LDS (one barrier per slab);
// fragments packed per CDNA5 ISA 7.12.2. Staging: float4 global loads ->
// v4h packed LDS stores, register-prefetched two slabs ahead so global
// latency and LDS staging overlap WMMA compute. Requires K % 32 == 0.
// VECB (compile-time): B rows contiguous in k (float4 path) vs strided scalar.
// ---------------------------------------------------------------------------
template <bool VECB>
__global__ __launch_bounds__(256)
void gemm_wmma_kernel(const float* __restrict__ A, const float* __restrict__ Bw,
                      const float* __restrict__ bias, const float* __restrict__ resid,
                      float* __restrict__ Out,
                      int M, int N, int K,
                      long sAm, long sBn, long sBk, long sOm, long sOn,
                      int NHd,
                      long zAb, long zAh, long zBb, long zBh, long zOb, long zOh,
                      int act) {
    __shared__ _Float16 As[2][128 * 32];   // [buf][m][k]
    __shared__ _Float16 Bs[2][64 * 32];    // [buf][n][k]

    const int tid  = threadIdx.x;
    const int wave = tid >> 5;
    const int lane = tid & 31;
    const int half = lane >> 4;
    const int lm   = lane & 15;

    const int m0 = blockIdx.y * 128;
    const int n0 = blockIdx.x * 64;
    const int z  = blockIdx.z;
    const int bq = z / NHd;
    const int hq = z % NHd;

    const float* Ab = A  + (long)bq * zAb + (long)hq * zAh;
    const float* Bb = Bw + (long)bq * zBb + (long)hq * zBh;
    float*       Ob = Out + (long)bq * zOb + (long)hq * zOh;
    const float* Rb = resid ? (resid + (long)bq * zOb + (long)hq * zOh) : nullptr;

    float4 ra[4];                       // A slab: 4 x float4 per thread
    float4 rbv[VECB ? 2 : 1];           // B slab (vector path)
    float  rbs[VECB ? 1 : 8];           // B slab (strided path)

    // ---- register-load helpers (branchless, clamped) ----
    auto loadA = [&](int k0) {
        #pragma unroll
        for (int u = 0; u < 4; ++u) {
            int g   = tid + u * 256;          // 0..1023
            int mm  = g >> 3;
            int kk4 = (g & 7) * 4;
            int gm  = m0 + mm;
            int gmc = gm < M ? gm : M - 1;
            float4 v = *(const float4*)(Ab + (long)gmc * sAm + (k0 + kk4));
            if (gm >= M) v = make_float4(0.f, 0.f, 0.f, 0.f);
            ra[u] = v;
        }
    };
    auto loadB = [&](int k0) {
        if constexpr (VECB) {
            #pragma unroll
            for (int u = 0; u < 2; ++u) {
                int g   = tid + u * 256;      // 0..511
                int nn  = g >> 3;
                int kk4 = (g & 7) * 4;
                int gn  = n0 + nn;
                int gnc = gn < N ? gn : N - 1;
                float4 v = *(const float4*)(Bb + (long)gnc * sBn + (k0 + kk4));
                if (gn >= N) v = make_float4(0.f, 0.f, 0.f, 0.f);
                rbv[u] = v;
            }
        } else {
            #pragma unroll
            for (int u = 0; u < 8; ++u) {
                int g  = tid + u * 256;       // 0..2047
                int nn = g >> 5;
                int kk = g & 31;
                int gn = n0 + nn;
                int gnc = gn < N ? gn : N - 1;
                float v = Bb[(long)gnc * sBn + (long)(k0 + kk) * sBk];
                rbs[u] = (gn < N) ? v : 0.0f;
            }
        }
    };
    auto stageLDS = [&](int buf) {
        #pragma unroll
        for (int u = 0; u < 4; ++u) {
            int g   = tid + u * 256;
            int mm  = g >> 3;
            int kk4 = (g & 7) * 4;
            float4 v = ra[u];
            v4h h = {(_Float16)v.x, (_Float16)v.y, (_Float16)v.z, (_Float16)v.w};
            *(v4h*)(&As[buf][mm * 32 + kk4]) = h;
        }
        if constexpr (VECB) {
            #pragma unroll
            for (int u = 0; u < 2; ++u) {
                int g   = tid + u * 256;
                int nn  = g >> 3;
                int kk4 = (g & 7) * 4;
                float4 v = rbv[u];
                v4h h = {(_Float16)v.x, (_Float16)v.y, (_Float16)v.z, (_Float16)v.w};
                *(v4h*)(&Bs[buf][nn * 32 + kk4]) = h;
            }
        } else {
            #pragma unroll
            for (int u = 0; u < 8; ++u) {
                int g  = tid + u * 256;
                int nn = g >> 5;
                int kk = g & 31;
                Bs[buf][nn * 32 + kk] = (_Float16)rbs[u];
            }
        }
    };

    v8f acc[4] = {};

    auto compute = [&](const _Float16* Asb, const _Float16* Bsb) {
        // A fragment (16x32 f16, ISA 7.12.2)
        v16h afrag;
        const _Float16* Ap = &Asb[(wave * 16 + lm) * 32];
        #pragma unroll
        for (int j = 0; j < 4; ++j) {
            int pk = half * 8 + 2 * j;
            afrag[2 * j]     = Ap[pk];
            afrag[2 * j + 1] = Ap[pk + 1];
        }
        #pragma unroll
        for (int j = 0; j < 4; ++j) {
            int pk = 16 + half * 8 + 2 * j;
            afrag[8 + 2 * j]     = Ap[pk];
            afrag[8 + 2 * j + 1] = Ap[pk + 1];
        }
        // 4 N-subtiles: B fragment 32x16
        #pragma unroll
        for (int s = 0; s < 4; ++s) {
            v16h bfrag;
            const _Float16* Bp = &Bsb[(s * 16 + lm) * 32];
            #pragma unroll
            for (int j = 0; j < 8; ++j) {
                int pk = half * 8 + 2 * j + (j >= 4 ? 8 : 0);
                bfrag[2 * j]     = Bp[pk];
                bfrag[2 * j + 1] = Bp[pk + 1];
            }
            acc[s] = __builtin_amdgcn_wmma_f32_16x16x32_f16(
                false, afrag, false, bfrag, (short)0, acc[s], false, false);
        }
    };

    // Prologue: stage slab 0 into buf0, prefetch slab 1 into regs.
    loadA(0); loadB(0);
    stageLDS(0);
    if (32 < K) { loadA(32); loadB(32); }
    __syncthreads();

    // Main loop, 2 slabs per trip (K % 64 == 0 for all launches).
    for (int k0 = 0; k0 < K; k0 += 64) {
        // phase 0: compute buf0 (slab k0), stage buf1 (slab k0+32)
        if (k0 + 32 < K) {
            stageLDS(1);
            if (k0 + 64 < K) { loadA(k0 + 64); loadB(k0 + 64); }
        }
        compute(As[0], Bs[0]);
        __syncthreads();
        if (k0 + 32 >= K) break;
        // phase 1: compute buf1 (slab k0+32), stage buf0 (slab k0+64)
        if (k0 + 64 < K) {
            stageLDS(0);
            if (k0 + 96 < K) { loadA(k0 + 96); loadB(k0 + 96); }
        }
        compute(As[1], Bs[1]);
        __syncthreads();
    }

    // Epilogue: C/D layout: VGPR r -> M = r + 8*half (+wave*16), N = lane%16
    const int mw = m0 + wave * 16;
    const bool fullM = (mw + 16) <= M;
    #pragma unroll
    for (int s = 0; s < 4; ++s) {
        int gn = n0 + s * 16 + lm;
        if (gn >= N) continue;              // never taken with our shapes
        float bv = bias ? bias[gn] : 0.0f;
        if (fullM) {
            #pragma unroll
            for (int r = 0; r < 8; ++r) {
                int gm = mw + half * 8 + r;
                long oidx = (long)gm * sOm + (long)gn * sOn;
                float v = acc[s][r] + bv;
                if (Rb)       v += Rb[oidx];
                if (act == 1) v = gelu_f(v);
                Ob[oidx] = v;
            }
        } else if (mw < M) {
            #pragma unroll
            for (int r = 0; r < 8; ++r) {
                int gm = mw + half * 8 + r;
                if (gm >= M) continue;
                long oidx = (long)gm * sOm + (long)gn * sOn;
                float v = acc[s][r] + bv;
                if (Rb)       v += Rb[oidx];
                if (act == 1) v = gelu_f(v);
                Ob[oidx] = v;
            }
        }
    }
}

// ---- LayerNorm over C=256; one wave per row ------------------------------
__global__ __launch_bounds__(256)
void ln_kernel(const float* __restrict__ x, const float* __restrict__ g,
               const float* __restrict__ bta, float* __restrict__ out) {
    int wave = threadIdx.x >> 5, lane = threadIdx.x & 31;
    long row = (long)blockIdx.x * 8 + wave;
    const float4* xr = (const float4*)(x + row * CC);
    float4 a = xr[lane], b = xr[lane + 32];
    float s  = a.x + a.y + a.z + a.w + b.x + b.y + b.z + b.w;
    float sq = a.x*a.x + a.y*a.y + a.z*a.z + a.w*a.w +
               b.x*b.x + b.y*b.y + b.z*b.z + b.w*b.w;
    s  = wave_sum(s);
    sq = wave_sum(sq);
    float m   = s * (1.0f / CC);
    float var = sq * (1.0f / CC) - m * m;
    float inv = rsqrtf(var + 1e-5f);
    const float4* gv = (const float4*)g;
    const float4* bv = (const float4*)bta;
    float4 g0 = gv[lane], g1 = gv[lane + 32];
    float4 b0 = bv[lane], b1 = bv[lane + 32];
    float4 o0, o1;
    o0.x = (a.x - m) * inv * g0.x + b0.x;  o0.y = (a.y - m) * inv * g0.y + b0.y;
    o0.z = (a.z - m) * inv * g0.z + b0.z;  o0.w = (a.w - m) * inv * g0.w + b0.w;
    o1.x = (b.x - m) * inv * g1.x + b1.x;  o1.y = (b.y - m) * inv * g1.y + b1.y;
    o1.z = (b.z - m) * inv * g1.z + b1.z;  o1.w = (b.w - m) * inv * g1.w + b1.w;
    float4* orow = (float4*)(out + row * CC);
    orow[lane]      = o0;
    orow[lane + 32] = o1;
}

// ---- k softmax over 4096 tokens per (b, ch); writes channel-major --------
__global__ __launch_bounds__(256)
void ksoftmax_kernel(const float* __restrict__ kin, float* __restrict__ kout) {
    __shared__ float red[256];
    int b = blockIdx.x >> 8, ch = blockIdx.x & 255;
    const float* base = kin + (long)b * NT * CC + ch;
    float vals[16];
    float mx = -1e30f;
    #pragma unroll
    for (int j = 0; j < 16; ++j) {
        float v = base[(long)(threadIdx.x + j * 256) * CC];
        vals[j] = v;
        mx = fmaxf(mx, v);
    }
    red[threadIdx.x] = mx; __syncthreads();
    for (int sft = 128; sft > 0; sft >>= 1) {
        if (threadIdx.x < sft) red[threadIdx.x] = fmaxf(red[threadIdx.x], red[threadIdx.x + sft]);
        __syncthreads();
    }
    mx = red[0]; __syncthreads();
    float s = 0.0f;
    #pragma unroll
    for (int j = 0; j < 16; ++j) { vals[j] = expf(vals[j] - mx); s += vals[j]; }
    red[threadIdx.x] = s; __syncthreads();
    for (int sft = 128; sft > 0; sft >>= 1) {
        if (threadIdx.x < sft) red[threadIdx.x] += red[threadIdx.x + sft];
        __syncthreads();
    }
    float inv = 1.0f / red[0];
    float* outb = kout + ((long)b * CC + ch) * NT;
    #pragma unroll
    for (int j = 0; j < 16; ++j) outb[threadIdx.x + j * 256] = vals[j] * inv;
}

// ---- q softmax over 64 head-channels, in place; one wave per (b,t,h) -----
__global__ __launch_bounds__(256)
void qsoftmax_kernel(float* __restrict__ q) {
    int wave = threadIdx.x >> 5, lane = threadIdx.x & 31;
    long row = (long)blockIdx.x * 8 + wave;       // (b*NT + t)*4 + h
    float2* p = (float2*)(q + row * 64);
    float2 v = p[lane];
    float mx = wave_max(fmaxf(v.x, v.y));
    float e0 = expf(v.x - mx), e1 = expf(v.y - mx);
    float s = wave_sum(e0 + e1);
    float inv = 1.0f / s;
    p[lane] = make_float2(e0 * inv, e1 * inv);
}

// ---- depthwise 3x3x3 conv (channel-last) + bias + exact GELU -------------
__global__ __launch_bounds__(256)
void dwconv_gelu_kernel(const float* __restrict__ in, const float* __restrict__ w,
                        const float* __restrict__ bias, float* __restrict__ out) {
    int t = blockIdx.x & (NT - 1);
    int b = blockIdx.x >> 12;
    int c = blockIdx.y * 256 + threadIdx.x;
    int d = t >> 8, h = (t >> 4) & 15, x = t & 15;
    const float* wc = w + c * 27;
    float acc = bias[c];
    #pragma unroll
    for (int dz = -1; dz <= 1; ++dz) {
        int zz = d + dz; if (zz < 0 || zz > 15) continue;
        #pragma unroll
        for (int dy = -1; dy <= 1; ++dy) {
            int yy = h + dy; if (yy < 0 || yy > 15) continue;
            #pragma unroll
            for (int dx = -1; dx <= 1; ++dx) {
                int xx = x + dx; if (xx < 0 || xx > 15) continue;
                long tt = ((long)b * NT + ((zz * 16 + yy) * 16 + xx));
                acc += in[tt * C2C + c] * wc[((dz + 1) * 3 + (dy + 1)) * 3 + (dx + 1)];
            }
        }
    }
    out[((long)b * NT + t) * C2C + c] = gelu_f(acc);
}

// ---- partial token-mean of n3: ypart[b][p][c] = sum over 256 tokens ------
__global__ __launch_bounds__(256)
void meanpart_kernel(const float* __restrict__ n3, float* __restrict__ ypart) {
    int b = blockIdx.x >> 4, p = blockIdx.x & 15;
    int c = threadIdx.x;
    const float* base = n3 + ((long)b * NT + p * 256) * CC + c;
    float acc = 0.0f;
    for (int j = 0; j < 256; ++j) acc += base[(long)j * CC];
    ypart[(long)blockIdx.x * CC + c] = acc;
}

// ---- channel attention: s = sigmoid(relu(y @ w1^T) @ w2^T) ---------------
__global__ __launch_bounds__(256)
void ca_kernel(const float* __restrict__ ypart, const float* __restrict__ w1,
               const float* __restrict__ w2, float* __restrict__ s) {
    __shared__ float ys[256];
    __shared__ float hs[16];
    int b = blockIdx.x, c = threadIdx.x;
    float acc = 0.0f;
    #pragma unroll
    for (int p = 0; p < 16; ++p) acc += ypart[((long)b * 16 + p) * CC + c];
    ys[c] = acc * (1.0f / NT);
    __syncthreads();
    if (c < 16) {
        float a = 0.0f;
        for (int j = 0; j < 256; ++j) a += ys[j] * w1[c * 256 + j];
        hs[c] = fmaxf(a, 0.0f);
    }
    __syncthreads();
    float a2 = 0.0f;
    #pragma unroll
    for (int j = 0; j < 16; ++j) a2 += hs[j] * w2[c * 16 + j];
    s[(long)b * CC + c] = 1.0f / (1.0f + expf(-a2));
}

// ---- add3 = add2 + n3 * s[b,c] -------------------------------------------
__global__ __launch_bounds__(256)
void add3_kernel(const float* __restrict__ add2, const float* __restrict__ n3,
                 const float* __restrict__ s, float* __restrict__ out) {
    long i = (long)blockIdx.x * 256 + threadIdx.x;
    int  c = (int)(i & (CC - 1));
    long b = i >> 20;                  // NT*CC = 2^20
    out[i] = add2[i] + n3[i] * s[b * CC + c];
}

// ---------------------------------------------------------------------------
static inline void gemm(hipStream_t st, const float* A, const float* Bw,
                        const float* bias, const float* resid, float* Out,
                        int M, int N, int K,
                        long sAm, long sBn, long sBk, long sOm, long sOn,
                        int Z, int NHd,
                        long zAb, long zAh, long zBb, long zBh, long zOb, long zOh,
                        int act) {
    dim3 grid((N + 63) / 64, (M + 127) / 128, Z);
    if (sBk == 1) {
        gemm_wmma_kernel<true><<<grid, 256, 0, st>>>(A, Bw, bias, resid, Out, M, N, K,
                                                     sAm, sBn, sBk, sOm, sOn, NHd,
                                                     zAb, zAh, zBb, zBh, zOb, zOh, act);
    } else {
        gemm_wmma_kernel<false><<<grid, 256, 0, st>>>(A, Bw, bias, resid, Out, M, N, K,
                                                      sAm, sBn, sBk, sOm, sOn, NHd,
                                                      zAb, zAh, zBb, zBh, zOb, zOh, act);
    }
}
static inline void gemm_std(hipStream_t st, const float* A, const float* Bw,
                            const float* bias, const float* resid, float* Out,
                            int M, int N, int K, int act) {
    gemm(st, A, Bw, bias, resid, Out, M, N, K,
         (long)K, (long)K, 1, (long)N, 1, 1, 1, 0, 0, 0, 0, 0, 0, act);
}

extern "C" void kernel_launch(void* const* d_in, const int* in_sizes, int n_in,
                              void* d_out, int out_size, void* d_ws, size_t ws_size,
                              hipStream_t stream) {
    const float* x       = (const float*)d_in[0];
    const float* n1g     = (const float*)d_in[4];   const float* n1b = (const float*)d_in[5];
    const float* n2g     = (const float*)d_in[6];   const float* n2b = (const float*)d_in[7];
    const float* n3g     = (const float*)d_in[8];   const float* n3b = (const float*)d_in[9];
    const float* n4g     = (const float*)d_in[10];  const float* n4b = (const float*)d_in[11];
    const float* Wk      = (const float*)d_in[12];  const float* bk  = (const float*)d_in[13];
    const float* Wq      = (const float*)d_in[14];  const float* bq  = (const float*)d_in[15];
    const float* Wv      = (const float*)d_in[16];  const float* bv  = (const float*)d_in[17];
    const float* Wr      = (const float*)d_in[18];  const float* br  = (const float*)d_in[19];
    const float* ca_w1   = (const float*)d_in[20];  const float* ca_w2 = (const float*)d_in[21];
    const float* m1_fc1w = (const float*)d_in[22];  const float* m1_fc1b = (const float*)d_in[23];
    const float* m1_dww  = (const float*)d_in[24];  const float* m1_dwb  = (const float*)d_in[25];
    const float* m1_fc2w = (const float*)d_in[26];  const float* m1_fc2b = (const float*)d_in[27];
    const float* m2_fc1w = (const float*)d_in[28];  const float* m2_fc1b = (const float*)d_in[29];
    const float* m2_dww  = (const float*)d_in[30];  const float* m2_dwb  = (const float*)d_in[31];
    const float* m2_fc2w = (const float*)d_in[32];  const float* m2_fc2b = (const float*)d_in[33];
    float* out = (float*)d_out;

    const size_t MC = (size_t)MTOT * CC;      // 8.39M floats
    const size_t MH = (size_t)MTOT * C2C;     // 33.55M floats
    float* ws = (float*)d_ws;
    size_t off = 0;
    float* bufA  = ws + off; off += MC;       // LN outputs (t1/t2/n3/t4)
    float* bufB  = ws + off; off += MC;       // k_lin -> attnT -> add2
    float* bufC  = ws + off; off += MC;       // q_lin (in-place softmax) -> add3
    float* bufD  = ws + off; off += MC;       // v_lin -> add1
    float* bufE  = ws + off; off += MC;       // ks_t (channel-major softmaxed k)
    float* hbig1 = ws + off; off += MH;       // fc1 out
    float* hbig2 = ws + off; off += MH;       // dwconv+gelu out
    float* ctxb  = ws + off; off += (size_t)32 * 64 * 64;
    float* ypart = ws + off; off += (size_t)BB * 16 * CC;
    float* sbuf  = ws + off; off += (size_t)BB * CC;
    (void)off; (void)ws_size; (void)in_sizes; (void)n_in; (void)out_size;

    const long SBH = (long)NT * CC;           // per-batch stride in token-major bufs

    // ---- efficient attention ----
    ln_kernel<<<MTOT / 8, 256, 0, stream>>>(x, n1g, n1b, bufA);
    gemm_std(stream, bufA, Wk, bk, nullptr, bufB, MTOT, CC, CC, 0);
    gemm_std(stream, bufA, Wq, bq, nullptr, bufC, MTOT, CC, CC, 0);
    gemm_std(stream, bufA, Wv, bv, nullptr, bufD, MTOT, CC, CC, 0);
    ksoftmax_kernel<<<BB * CC, 256, 0, stream>>>(bufB, bufE);
    qsoftmax_kernel<<<(BB * NT * 4) / 8, 256, 0, stream>>>(bufC);
    // ctx[b,h][kc,vc] = sum_t ks[b,h,kc,t] * v[b,t,h*64+vc]; stored transposed [vc][kc]
    gemm(stream, bufE, bufD, nullptr, nullptr, ctxb,
         64, 64, NT,
         /*sAm*/ NT, /*sBn*/ 1, /*sBk*/ CC, /*sOm*/ 1, /*sOn*/ 64,
         /*Z*/ 32, /*NHd*/ 4,
         /*zAb*/ (long)CC * NT, /*zAh*/ 64L * NT,
         /*zBb*/ SBH,           /*zBh*/ 64,
         /*zOb*/ 4L * 64 * 64,  /*zOh*/ 64L * 64, 0);
    // o[b][t][h*64+vc] = sum_kc ctxT[vc,kc] * q[b,t,h*64+kc]  (token-major output)
    gemm(stream, ctxb, bufC, nullptr, nullptr, bufB,
         64, NT, 64,
         /*sAm*/ 64, /*sBn*/ CC, /*sBk*/ 1, /*sOm*/ 1, /*sOn*/ CC,
         /*Z*/ 32, /*NHd*/ 4,
         /*zAb*/ 4L * 64 * 64, /*zAh*/ 64L * 64,
         /*zBb*/ SBH,          /*zBh*/ 64,
         /*zOb*/ SBH,          /*zOh*/ 64, 0);
    gemm_std(stream, bufB, Wr, br, x, bufD, MTOT, CC, 256, 0);          // add1

    // ---- MixFFN 1 ----
    ln_kernel<<<MTOT / 8, 256, 0, stream>>>(bufD, n2g, n2b, bufA);
    gemm_std(stream, bufA, m1_fc1w, m1_fc1b, nullptr, hbig1, MTOT, C2C, CC, 0);
    dwconv_gelu_kernel<<<dim3(BB * NT, C2C / 256), 256, 0, stream>>>(hbig1, m1_dww, m1_dwb, hbig2);
    gemm_std(stream, hbig2, m1_fc2w, m1_fc2b, bufD, bufB, MTOT, CC, C2C, 0);  // add2

    // ---- channel attention ----
    ln_kernel<<<MTOT / 8, 256, 0, stream>>>(bufB, n3g, n3b, bufA);      // n3
    meanpart_kernel<<<BB * 16, 256, 0, stream>>>(bufA, ypart);
    ca_kernel<<<BB, 256, 0, stream>>>(ypart, ca_w1, ca_w2, sbuf);
    add3_kernel<<<(unsigned)(MC / 256), 256, 0, stream>>>(bufB, bufA, sbuf, bufC);  // add3

    // ---- MixFFN 2 ----
    ln_kernel<<<MTOT / 8, 256, 0, stream>>>(bufC, n4g, n4b, bufA);
    gemm_std(stream, bufA, m2_fc1w, m2_fc1b, nullptr, hbig1, MTOT, C2C, CC, 0);
    dwconv_gelu_kernel<<<dim3(BB * NT, C2C / 256), 256, 0, stream>>>(hbig1, m2_dww, m2_dwb, hbig2);
    gemm_std(stream, hbig2, m2_fc2w, m2_fc2b, bufC, out, MTOT, CC, C2C, 0);
}
